// SpikformerPrune_38173669327030
// MI455X (gfx1250) — matrix operand order repaired and checked by reading, not compiled, not run on hardware
//
#include <hip/hip_runtime.h>
#include <hip/hip_bf16.h>

// ---------------------------------------------------------------------------
// Spikformer w/ token pruning, MI455X (gfx1250, wave32, WMMA f16 16x16x32).
// All GEMM-shaped compute (convs via 9-tap accumulation, q/k/v/proj/fc1/fc2,
// both attention einsums, predictor) runs through v_wmma_f32_16x16x32_f16.
// Spikes are {0,1} -> exact in f16; S=qk^T*0.125 is multiples of 0.125 -> exact
// in f16; head dim 32 == WMMA K. 2x2 register tiling: 4 independent WMMAs per
// K-step sharing A/B fragments (2 loads per WMMA, back-to-back XDL issue).
// ---------------------------------------------------------------------------

typedef __attribute__((ext_vector_type(16))) _Float16 v16h;
typedef __attribute__((ext_vector_type(8)))  _Float16 v8h;
typedef __attribute__((ext_vector_type(8)))  float    v8f;

#define CAT16(lo,hi) __builtin_shufflevector((lo),(hi),0,1,2,3,4,5,6,7,8,9,10,11,12,13,14,15)
#define WMMA32(a,b,c) __builtin_amdgcn_wmma_f32_16x16x32_f16(false,(a),false,(b),(short)0,(c),false,false)

__device__ __forceinline__ float lif_step(float& vm, float xv, float vth) {
  vm += (xv - vm) * 0.5f;              // tau = 2
  float s = (vm >= vth) ? 1.f : 0.f;   // spike(v - vth)
  vm *= (1.f - s);                     // hard reset
  return s;
}

__device__ __forceinline__ float gelu_tanh(float x) {
  return 0.5f * x * (1.f + tanhf(0.7978845608028654f * (x + 0.044715f * x * x * x)));
}

// ---------------------------------------------------------------------------
// Generic GEMM: C[M x Nn] = A[M x K] * B[K x Nn] (+bias) * alpha
// A row-major f16 (lda), B given TRANSPOSED row-major f16 (BT[Nn x K], ldb).
// 4 waves/block; wave owns a 32x32 output tile (2x2 of 16x16 WMMA tiles).
// K multiple of 32, M and Nn multiples of 32.
// ---------------------------------------------------------------------------
__global__ void gemm_bt_kernel(const _Float16* __restrict__ A, long long sA, int lda,
                               const _Float16* __restrict__ BT, long long sB, int ldb,
                               void* __restrict__ Cv, long long sC, int ldc,
                               const float* __restrict__ bias, float alpha,
                               int M, int K, int out16)
{
  int lane = threadIdx.x & 31;
  int wave = threadIdx.x >> 5;
  int row_base = blockIdx.y * 128 + wave * 32;
  if (row_base >= M) return;
  int z = blockIdx.z;
  A  += (long long)z * sA;
  BT += (long long)z * sB;

  int m0 = row_base + (lane & 15);
  int n0 = blockIdx.x * 32 + (lane & 15);
  const _Float16* ap0 = A  + (size_t)m0 * lda;
  const _Float16* ap1 = ap0 + (size_t)16 * lda;
  const _Float16* bp0 = BT + (size_t)n0 * ldb;
  const _Float16* bp1 = bp0 + (size_t)16 * ldb;
  // A 16x32 layout (matches B): lanes 0-15 hold K0..15, 16-31 K16..31
  int ka0 = (lane < 16) ? 0 : 16;
  int ka1 = (lane < 16) ? 8 : 24;
  int kb  = (lane < 16) ? 0 : 16;   // B 32x16: lanes 0-15 K0..15, 16-31 K16..31

  v8f acc00 = {}, acc01 = {}, acc10 = {}, acc11 = {};
  for (int k0 = 0; k0 < K; k0 += 32) {
    if (k0 + 64 < K) {
      __builtin_prefetch(ap0 + k0 + 64, 0, 1);
      __builtin_prefetch(bp0 + k0 + 64, 0, 1);
    }
    v16h a0 = CAT16(*(const v8h*)(ap0 + k0 + ka0), *(const v8h*)(ap0 + k0 + ka1));
    v16h a1 = CAT16(*(const v8h*)(ap1 + k0 + ka0), *(const v8h*)(ap1 + k0 + ka1));
    v16h b0 = CAT16(*(const v8h*)(bp0 + k0 + kb),  *(const v8h*)(bp0 + k0 + kb + 8));
    v16h b1 = CAT16(*(const v8h*)(bp1 + k0 + kb),  *(const v8h*)(bp1 + k0 + kb + 8));
    acc00 = WMMA32(a0, b0, acc00);
    acc01 = WMMA32(a0, b1, acc01);
    acc10 = WMMA32(a1, b0, acc10);
    acc11 = WMMA32(a1, b1, acc11);
  }
  float bv0 = bias ? bias[n0]      : 0.f;
  float bv1 = bias ? bias[n0 + 16] : 0.f;
  int rb = row_base + ((lane < 16) ? 0 : 8);
  if (out16) {
    _Float16* C = (_Float16*)Cv + (long long)z * sC;
#pragma unroll
    for (int i = 0; i < 8; ++i) {
      C[(size_t)(rb + i)      * ldc + n0]      = (_Float16)((acc00[i] + bv0) * alpha);
      C[(size_t)(rb + i)      * ldc + n0 + 16] = (_Float16)((acc01[i] + bv1) * alpha);
      C[(size_t)(rb + 16 + i) * ldc + n0]      = (_Float16)((acc10[i] + bv0) * alpha);
      C[(size_t)(rb + 16 + i) * ldc + n0 + 16] = (_Float16)((acc11[i] + bv1) * alpha);
    }
  } else {
    float* C = (float*)Cv + (long long)z * sC;
#pragma unroll
    for (int i = 0; i < 8; ++i) {
      C[(size_t)(rb + i)      * ldc + n0]      = (acc00[i] + bv0) * alpha;
      C[(size_t)(rb + i)      * ldc + n0 + 16] = (acc01[i] + bv1) * alpha;
      C[(size_t)(rb + 16 + i) * ldc + n0]      = (acc10[i] + bv0) * alpha;
      C[(size_t)(rb + 16 + i) * ldc + n0 + 16] = (acc11[i] + bv1) * alpha;
    }
  }
}

// ---------------------------------------------------------------------------
// Conv3x3 (stride 1, pad 1) as 9 shifted-tap WMMA GEMMs over zero-padded
// NHWC f16 input. wpk[tap][Cout][Cin] f16 (BT form per tap). 2x2 tiling:
// wave owns 32 pixels x 32 output channels. Cout multiple of 32.
// ---------------------------------------------------------------------------
__global__ void conv3x3_wmma_kernel(const _Float16* __restrict__ in,
                                    const _Float16* __restrict__ wpk,
                                    float* __restrict__ out,
                                    int Hh, int Cin, int Cout, int M, int logHW, int logW)
{
  int lane = threadIdx.x & 31;
  int wave = threadIdx.x >> 5;
  int row_base = blockIdx.y * 128 + wave * 32;
  if (row_base >= M) return;
  int ncol = blockIdx.x * 32 + (lane & 15);
  int m0 = row_base + (lane & 15);
  int m1 = m0 + 16;
  int Wp2 = (1 << logW) + 2;
  int hwm = (1 << logHW) - 1, wm = (1 << logW) - 1;
  int img0 = m0 >> logHW, p0 = m0 & hwm;
  int img1 = m1 >> logHW, p1 = m1 & hwm;
  size_t pixbase0 = ((size_t)img0 * (Hh + 2) + (p0 >> logW)) * Wp2 + (p0 & wm);
  size_t pixbase1 = ((size_t)img1 * (Hh + 2) + (p1 >> logW)) * Wp2 + (p1 & wm);
  int ka0 = (lane < 16) ? 0 : 16;
  int ka1 = (lane < 16) ? 8 : 24;
  int kb  = (lane < 16) ? 0 : 16;
  const _Float16* bcol0 = wpk + (size_t)ncol * Cin;
  const _Float16* bcol1 = bcol0 + (size_t)16 * Cin;

  v8f acc00 = {}, acc01 = {}, acc10 = {}, acc11 = {};
#pragma unroll
  for (int tap = 0; tap < 9; ++tap) {
    size_t toff = (size_t)(tap / 3) * Wp2 + (tap % 3);
    const _Float16* ap0 = in + (pixbase0 + toff) * Cin;
    const _Float16* ap1 = in + (pixbase1 + toff) * Cin;
    const _Float16* bp0 = bcol0 + (size_t)tap * Cout * Cin;
    const _Float16* bp1 = bcol1 + (size_t)tap * Cout * Cin;
    for (int k0 = 0; k0 < Cin; k0 += 32) {
      v16h a0 = CAT16(*(const v8h*)(ap0 + k0 + ka0), *(const v8h*)(ap0 + k0 + ka1));
      v16h a1 = CAT16(*(const v8h*)(ap1 + k0 + ka0), *(const v8h*)(ap1 + k0 + ka1));
      v16h b0 = CAT16(*(const v8h*)(bp0 + k0 + kb),  *(const v8h*)(bp0 + k0 + kb + 8));
      v16h b1 = CAT16(*(const v8h*)(bp1 + k0 + kb),  *(const v8h*)(bp1 + k0 + kb + 8));
      acc00 = WMMA32(a0, b0, acc00);
      acc01 = WMMA32(a0, b1, acc01);
      acc10 = WMMA32(a1, b0, acc10);
      acc11 = WMMA32(a1, b1, acc11);
    }
  }
  int rb = row_base + ((lane < 16) ? 0 : 8);
#pragma unroll
  for (int i = 0; i < 8; ++i) {
    out[(size_t)(rb + i)      * Cout + ncol]      = acc00[i];
    out[(size_t)(rb + i)      * Cout + ncol + 16] = acc01[i];
    out[(size_t)(rb + 16 + i) * Cout + ncol]      = acc10[i];
    out[(size_t)(rb + 16 + i) * Cout + ncol + 16] = acc11[i];
  }
}

// --------------------------- weight preparation ----------------------------
__global__ void transpose_f16_kernel(const float* __restrict__ src, _Float16* __restrict__ dst,
                                     int K, int N) {  // src[K][N] -> dst[N][K]
  int id = blockIdx.x * 256 + threadIdx.x;
  if (id >= K * N) return;
  int n = id / K, kk = id % K;
  dst[id] = (_Float16)src[(size_t)kk * N + n];
}

__global__ void convpack_kernel(const float* __restrict__ src, _Float16* __restrict__ dst,
                                int Cout, int Cin) {   // OIHW -> [tap][Cout][Cin]
  int id = blockIdx.x * 256 + threadIdx.x;
  if (id >= 9 * Cout * Cin) return;
  int tap = id / (Cout * Cin);
  int r = id % (Cout * Cin);
  int o = r / Cin, i = r % Cin;
  dst[id] = (_Float16)src[(size_t)(o * Cin + i) * 9 + tap];
}

// ------------------------------- SPS kernels -------------------------------
// conv0: Cin=2 direct (too small for WMMA K). x is f32 NCHW (T,B,2,128,128).
__global__ void conv0_kernel(const float* __restrict__ xin, const float* __restrict__ w,
                             float* __restrict__ out) {
  int id = blockIdx.x * 256 + threadIdx.x;
  if (id >= 8 * 16384 * 32) return;
  int co = id & 31;
  int p = (id >> 5) & 16383;
  int img = id >> 19;
  int y = p >> 7, x = p & 127;
  float acc = 0.f;
#pragma unroll
  for (int ci = 0; ci < 2; ++ci)
#pragma unroll
    for (int dy = 0; dy < 3; ++dy) {
      int yyy = y + dy - 1; if ((unsigned)yyy >= 128u) continue;
#pragma unroll
      for (int dx = 0; dx < 3; ++dx) {
        int xxx = x + dx - 1; if ((unsigned)xxx >= 128u) continue;
        acc += xin[((size_t)(img * 2 + ci) << 14) + (yyy << 7) + xxx] *
               w[((co * 2 + ci) * 3 + dy) * 3 + dx];
      }
    }
  out[id] = acc;
}

// BN + LIF(vth=1) over T, write f16 spikes into zero-padded NHWC buffer.
__global__ void bn_lif_pad_kernel(const float* __restrict__ gin,
                                  const float* __restrict__ g, const float* __restrict__ bb,
                                  const float* __restrict__ bm, const float* __restrict__ bv,
                                  _Float16* __restrict__ spad,
                                  int HW, int Ww, int Hh, int Cc, int total)
{
  int id = blockIdx.x * 256 + threadIdx.x;
  if (id >= total) return;
  int c = id % Cc;
  int rest = id / Cc;
  int p = rest % HW;
  int bi = rest / HW;
  int y = p / Ww, x = p % Ww;
  float sc = g[c] * rsqrtf(bv[c] + 1e-5f);
  float sh = bb[c] - bm[c] * sc;
  float vm = 0.f;
  long long tstride = (long long)2 * HW * Cc;
  size_t gbase = (size_t)(bi * HW + p) * Cc + c;
  for (int t = 0; t < 4; ++t) {
    float xv = gin[gbase + (size_t)t * tstride] * sc + sh;
    float s = lif_step(vm, xv, 1.0f);
    spad[(((size_t)(t * 2 + bi) * (Hh + 2) + (y + 1)) * (Ww + 2) + (x + 1)) * Cc + c] = (_Float16)s;
  }
}

// 3x3 stride-2 maxpool on padded f16 spikes (values >= 0, zero pad == -inf pad).
__global__ void pool_kernel(const _Float16* __restrict__ sin, _Float16* __restrict__ sout,
                            int Hi, int Cc, int total) {
  int id = blockIdx.x * 256 + threadIdx.x;
  if (id >= total) return;
  int Ho = Hi >> 1;
  int c = id % Cc;
  int rest = id / Cc;
  int xo = rest % Ho; rest /= Ho;
  int yo = rest % Ho;
  int img = rest / Ho;
  float mx = 0.f;
#pragma unroll
  for (int i = 0; i < 3; ++i)
#pragma unroll
    for (int j = 0; j < 3; ++j) {
      float v = (float)sin[(((size_t)img * (Hi + 2) + (2 * yo + i)) * (Hi + 2) + (2 * xo + j)) * Cc + c];
      mx = fmaxf(mx, v);
    }
  sout[(((size_t)img * (Ho + 2) + (yo + 1)) * (Ho + 2) + (xo + 1)) * Cc + c] = (_Float16)mx;
}

// RPE BN+LIF + residual with pre-RPE spikes -> token stream x (f32 + f16).
__global__ void lif_tokens_kernel(const float* __restrict__ gin,
                                  const float* __restrict__ g, const float* __restrict__ bb,
                                  const float* __restrict__ bm, const float* __restrict__ bv,
                                  const _Float16* __restrict__ s3p,
                                  float* __restrict__ xa, _Float16* __restrict__ xh)
{
  int id = blockIdx.x * 256 + threadIdx.x;
  if (id >= 2 * 1024 * 256) return;
  int c = id & 255;
  int n = (id >> 8) & 1023;
  int bi = id >> 18;
  int y = n >> 5, x = n & 31;
  float sc = g[c] * rsqrtf(bv[c] + 1e-5f);
  float sh = bb[c] - bm[c] * sc;
  float vm = 0.f;
  for (int t = 0; t < 4; ++t) {
    size_t r = (size_t)(t * 2 + bi) * 1024 + n;
    float xv = gin[r * 256 + c] * sc + sh;
    float s = lif_step(vm, xv, 1.0f);
    float feat = (float)s3p[(((size_t)(t * 2 + bi) * 34 + (y + 1)) * 34 + (x + 1)) * 256 + c];
    float val = s + feat;
    xa[r * 256 + c] = val;
    xh[r * 256 + c] = (_Float16)val;
  }
}

// ---------------------------- transformer kernels --------------------------
// q/k/v: BN+LIF -> spikes per head: [tb][h][n][32]; writeT=1 -> [tb][h][32][Np].
__global__ void lif_qkv_kernel(const float* __restrict__ gin,
                               const float* __restrict__ g, const float* __restrict__ bb,
                               const float* __restrict__ bm, const float* __restrict__ bv,
                               _Float16* __restrict__ out, int N, int Np, int writeT, int total)
{
  int id = blockIdx.x * 256 + threadIdx.x;
  if (id >= total) return;
  int c = id & 255;
  int rest = id >> 8;
  int n = rest % N;
  int bi = rest / N;
  int h = c >> 5, d = c & 31;
  float sc = g[c] * rsqrtf(bv[c] + 1e-5f);
  float sh = bb[c] - bm[c] * sc;
  float vm = 0.f;
  for (int t = 0; t < 4; ++t) {
    size_t r = (size_t)(t * 2 + bi) * Np + n;
    float xv = gin[r * 256 + c] * sc + sh;
    float s = lif_step(vm, xv, 1.0f);
    if (!writeT)
      out[((((size_t)(t * 2 + bi) * 8 + h) * Np + n) << 5) + d] = (_Float16)s;
    else
      out[(((size_t)(t * 2 + bi) * 8 + h) * 32 + d) * (size_t)Np + n] = (_Float16)s;
  }
}

// attention output LIF (vth=0.5), reassemble heads -> [tb*Np][256] f16 spikes.
__global__ void lif_attn_kernel(const float* __restrict__ yatt, _Float16* __restrict__ ysp,
                                int N, int Np, int total)
{
  int id = blockIdx.x * 256 + threadIdx.x;
  if (id >= total) return;
  int d = id & 31;
  int rest = id >> 5;
  int n = rest % N; rest /= N;
  int h = rest & 7;
  int bi = rest >> 3;
  float vm = 0.f;
  for (int t = 0; t < 4; ++t) {
    float yv = yatt[((((size_t)(t * 2 + bi) * 8 + h) * Np + n) << 5) + d];
    float s = lif_step(vm, yv, 0.5f);
    ysp[((size_t)(t * 2 + bi) * Np + n) * 256 + (h << 5) + d] = (_Float16)s;
  }
}

// BN+LIF + residual add into x (proj / fc2 epilogue).
__global__ void lif_res_kernel(const float* __restrict__ gin,
                               const float* __restrict__ g, const float* __restrict__ bb,
                               const float* __restrict__ bm, const float* __restrict__ bv,
                               float* __restrict__ xa, _Float16* __restrict__ xh,
                               int N, int Np, int total)
{
  int id = blockIdx.x * 256 + threadIdx.x;
  if (id >= total) return;
  int c = id & 255;
  int rest = id >> 8;
  int n = rest % N;
  int bi = rest / N;
  float sc = g[c] * rsqrtf(bv[c] + 1e-5f);
  float sh = bb[c] - bm[c] * sc;
  float vm = 0.f;
  for (int t = 0; t < 4; ++t) {
    size_t r = (size_t)(t * 2 + bi) * Np + n;
    float xv = gin[r * 256 + c] * sc + sh;
    float s = lif_step(vm, xv, 1.0f);
    float nv = xa[r * 256 + c] + s;
    xa[r * 256 + c] = nv;
    xh[r * 256 + c] = (_Float16)nv;
  }
}

// fc1 BN+LIF -> hidden spikes f16 [tb*Np][1024]
__global__ void lif_fc1_kernel(const float* __restrict__ gin,
                               const float* __restrict__ g, const float* __restrict__ bb,
                               const float* __restrict__ bm, const float* __restrict__ bv,
                               _Float16* __restrict__ hs, int N, int Np, int total)
{
  int id = blockIdx.x * 256 + threadIdx.x;
  if (id >= total) return;
  int c = id & 1023;
  int rest = id >> 10;
  int n = rest % N;
  int bi = rest / N;
  float sc = g[c] * rsqrtf(bv[c] + 1e-5f);
  float sh = bb[c] - bm[c] * sc;
  float vm = 0.f;
  for (int t = 0; t < 4; ++t) {
    size_t r = (size_t)(t * 2 + bi) * Np + n;
    float xv = gin[r * 1024 + c] * sc + sh;
    float s = lif_step(vm, xv, 1.0f);
    hs[r * 1024 + c] = (_Float16)s;
  }
}

// ----------------------------- pruning kernels -----------------------------
__global__ void mean_kernel(const float* __restrict__ xa, _Float16* __restrict__ xm,
                            int N, int Np, int total)
{
  int id = blockIdx.x * 256 + threadIdx.x;
  if (id >= total) return;
  int c = id & 255;
  int rest = id >> 8;
  int n = rest % N;
  int bi = rest / N;
  float a = 0.f;
  for (int t = 0; t < 4; ++t) a += xa[((size_t)(t * 2 + bi) * Np + n) * 256 + c];
  xm[((size_t)bi * Np + n) * 256 + c] = (_Float16)(a * 0.25f);
}

// score = z0 - z1 (same ranking as log_softmax[...,0]); gelu tanh approx.
__global__ void pred_score_kernel(const float* __restrict__ predh,
                                  const float* __restrict__ w2, const float* __restrict__ b2,
                                  float* __restrict__ score, int N, int Np, int total)
{
  int id = blockIdx.x * 256 + threadIdx.x;
  if (id >= total) return;
  int n = id % N;
  int bi = id / N;
  const float* hr = predh + ((size_t)bi * Np + n) * 128;
  float acc = b2[0] - b2[1];
  for (int j = 0; j < 128; ++j)
    acc += gelu_tanh(hr[j]) * (w2[2 * j] - w2[2 * j + 1]);
  score[bi * 1024 + n] = acc;
}

__global__ void topk_kernel(const float* __restrict__ score, int* __restrict__ idxo,
                            int N, int k)
{
  __shared__ float sc[1024];
  __shared__ float rv[256];
  __shared__ int   ri[256];
  int b = blockIdx.x, tid = threadIdx.x;
  for (int n = tid; n < 1024; n += 256) sc[n] = (n < N) ? score[b * 1024 + n] : -1e30f;
  __syncthreads();
  for (int j = 0; j < k; ++j) {
    float bv = -1e30f; int bi = -1;
    for (int n = tid; n < 1024; n += 256) { float v = sc[n]; if (v > bv) { bv = v; bi = n; } }
    rv[tid] = bv; ri[tid] = bi;
    __syncthreads();
    for (int s = 128; s > 0; s >>= 1) {
      if (tid < s) {
        bool take = (rv[tid + s] > rv[tid]) ||
                    (rv[tid + s] == rv[tid] && ri[tid + s] >= 0 && (ri[tid] < 0 || ri[tid + s] < ri[tid]));
        if (take) { rv[tid] = rv[tid + s]; ri[tid] = ri[tid + s]; }
      }
      __syncthreads();
    }
    if (tid == 0) { idxo[b * 1024 + j] = ri[0]; sc[ri[0]] = -1e30f; }
    __syncthreads();
  }
}

__global__ void gather_kernel(const float* __restrict__ xo, const int* __restrict__ idx,
                              float* __restrict__ xn, _Float16* __restrict__ xhn,
                              int Npo, int Npn, int k, int total)
{
  int id = blockIdx.x * 256 + threadIdx.x;
  if (id >= total) return;
  int c = id & 255;
  int rest = id >> 8;
  int j = rest % k; rest /= k;
  int bi = rest & 1;
  int t = rest >> 1;
  int n = idx[bi * 1024 + j];
  float v = xo[((size_t)(t * 2 + bi) * Npo + n) * 256 + c];
  size_t o = ((size_t)(t * 2 + bi) * Npn + j) * 256 + c;
  xn[o] = v;
  xhn[o] = (_Float16)v;
}

// ------------------------------- head -------------------------------------
__global__ void head_kernel(const float* __restrict__ xa, const float* __restrict__ hw,
                            const float* __restrict__ hb, float* __restrict__ out,
                            int N, int Np)
{
  __shared__ float xm[256];
  int b = blockIdx.x, tid = threadIdx.x;
  float a = 0.f;
  for (int t = 0; t < 4; ++t)
    for (int n = 0; n < N; ++n)
      a += xa[((size_t)(t * 2 + b) * Np + n) * 256 + tid];
  xm[tid] = a * (1.f / (4.f * (float)N));
  __syncthreads();
  if (tid < 11) {
    float acc = hb[tid];
    for (int c = 0; c < 256; ++c) acc += xm[c] * hw[c * 11 + tid];
    out[b * 11 + tid] = acc;
  }
}

// ===========================================================================
// Host side
// ===========================================================================
struct BNp { const float *b, *g, *m, *v; };
struct Linp { const float* bias; BNp bn; const float* w; };
struct Blkp { Linp fc1, fc2, k, proj, q, v; };
struct Predp { const float *b1, *b2, *w1, *w2; };

static inline dim3 eg(long long total) { return dim3((unsigned)((total + 255) / 256)); }

static inline void gemm(hipStream_t s,
                        const _Float16* A, long long sA, int lda,
                        const _Float16* BT, long long sB, int ldb,
                        void* C, long long sC, int ldc,
                        const float* bias, float alpha,
                        int M, int Nn, int K, int batch, int out16)
{
  dim3 g(Nn / 32, (M + 127) / 128, batch);
  gemm_bt_kernel<<<g, 128, 0, s>>>(A, sA, lda, BT, sB, ldb, C, sC, ldc, bias, alpha, M, K, out16);
}

extern "C" void kernel_launch(void* const* d_in, const int* in_sizes, int n_in,
                              void* d_out, int out_size, void* d_ws, size_t ws_size,
                              hipStream_t stream)
{
  (void)in_sizes; (void)n_in; (void)out_size;
  // ---- parse inputs (JAX pytree sorted-key flatten: params then x) ----
  int ip = 0;
  auto F = [&]() { return (const float*)d_in[ip++]; };
  auto getBN = [&]() { BNp r; r.b = F(); r.g = F(); r.m = F(); r.v = F(); return r; };
  auto getLin = [&]() { Linp l; l.bias = F(); l.bn = getBN(); l.w = F(); return l; };
  Blkp blk[4];
  for (int i = 0; i < 4; ++i) {
    blk[i].fc1 = getLin(); blk[i].fc2 = getLin(); blk[i].k = getLin();
    blk[i].proj = getLin(); blk[i].q = getLin(); blk[i].v = getLin();
  }
  const float* head_b = F();
  const float* head_w = F();
  Predp pr[3];
  for (int j = 0; j < 3; ++j) { pr[j].b1 = F(); pr[j].b2 = F(); pr[j].w1 = F(); pr[j].w2 = F(); }
  BNp bn0 = getBN(), bn1 = getBN(), bn2 = getBN(), bn3 = getBN();
  const float* c0w = F(); const float* c1w = F(); const float* c2w = F(); const float* c3w = F();
  BNp rpebn = getBN();
  const float* rpew = F();
  const float* xin = F();

  // ---- workspace layout ----
  char* wsb = (char*)d_ws;
  size_t off = 0;
  auto alloc = [&](size_t bytes) { void* p = wsb + off; off = (off + bytes + 255) & ~(size_t)255; return p; };

  float* conv_out = (float*)alloc((size_t)131072 * 128 * 4);
  _Float16* s0  = (_Float16*)alloc((size_t)8 * 130 * 130 * 32 * 2);
  _Float16* s1  = (_Float16*)alloc((size_t)8 * 130 * 130 * 64 * 2);
  _Float16* s2  = (_Float16*)alloc((size_t)8 * 130 * 130 * 128 * 2);
  _Float16* s2p = (_Float16*)alloc((size_t)8 * 66 * 66 * 128 * 2);
  _Float16* s3  = (_Float16*)alloc((size_t)8 * 66 * 66 * 256 * 2);
  _Float16* s3p = (_Float16*)alloc((size_t)8 * 34 * 34 * 256 * 2);
  float*    xA  = (float*)alloc((size_t)8 * 1024 * 256 * 4);
  float*    xB  = (float*)alloc((size_t)8 * 1024 * 256 * 4);
  _Float16* xhA = (_Float16*)alloc((size_t)8 * 1024 * 256 * 2);
  _Float16* xhB = (_Float16*)alloc((size_t)8 * 1024 * 256 * 2);
  float*    lin_out = (float*)alloc((size_t)8 * 1024 * 1024 * 4);
  _Float16* qs  = (_Float16*)alloc((size_t)8 * 8 * 1024 * 32 * 2);
  _Float16* ks  = (_Float16*)alloc((size_t)8 * 8 * 1024 * 32 * 2);
  _Float16* vT  = (_Float16*)alloc((size_t)8 * 8 * 32 * 1024 * 2);
  _Float16* S16 = (_Float16*)alloc((size_t)16 * 1024 * 1024 * 2);
  float*    yatt = (float*)alloc((size_t)8 * 8 * 1024 * 32 * 4);
  _Float16* ysp = (_Float16*)alloc((size_t)8 * 1024 * 256 * 2);
  _Float16* hsp = (_Float16*)alloc((size_t)8 * 1024 * 1024 * 2);
  _Float16* xm16 = (_Float16*)alloc((size_t)2 * 1024 * 256 * 2);
  float*    predh = (float*)alloc((size_t)2 * 1024 * 128 * 4);
  float*    score = (float*)alloc((size_t)2 * 1024 * 4);
  int*      idxb  = (int*)alloc((size_t)2 * 1024 * 4);
  _Float16 *wqT[4], *wkT[4], *wvT[4], *wpT[4], *wf1T[4], *wf2T[4], *wp1T[3];
  for (int i = 0; i < 4; ++i) {
    wqT[i]  = (_Float16*)alloc((size_t)256 * 256 * 2);
    wkT[i]  = (_Float16*)alloc((size_t)256 * 256 * 2);
    wvT[i]  = (_Float16*)alloc((size_t)256 * 256 * 2);
    wpT[i]  = (_Float16*)alloc((size_t)256 * 256 * 2);
    wf1T[i] = (_Float16*)alloc((size_t)1024 * 256 * 2);
    wf2T[i] = (_Float16*)alloc((size_t)256 * 1024 * 2);
  }
  for (int j = 0; j < 3; ++j) wp1T[j] = (_Float16*)alloc((size_t)128 * 256 * 2);
  _Float16* wpk1 = (_Float16*)alloc((size_t)9 * 64 * 32 * 2);
  _Float16* wpk2 = (_Float16*)alloc((size_t)9 * 128 * 64 * 2);
  _Float16* wpk3 = (_Float16*)alloc((size_t)9 * 256 * 128 * 2);
  _Float16* wpkR = (_Float16*)alloc((size_t)9 * 256 * 256 * 2);

  // zero whole workspace once per launch: establishes all zero pads (graph-safe)
  hipMemsetAsync(d_ws, 0, ws_size, stream);

  // ---- weight prep ----
  for (int i = 0; i < 4; ++i) {
    transpose_f16_kernel<<<eg(65536), 256, 0, stream>>>(blk[i].q.w,   wqT[i],  256, 256);
    transpose_f16_kernel<<<eg(65536), 256, 0, stream>>>(blk[i].k.w,   wkT[i],  256, 256);
    transpose_f16_kernel<<<eg(65536), 256, 0, stream>>>(blk[i].v.w,   wvT[i],  256, 256);
    transpose_f16_kernel<<<eg(65536), 256, 0, stream>>>(blk[i].proj.w, wpT[i], 256, 256);
    transpose_f16_kernel<<<eg(262144), 256, 0, stream>>>(blk[i].fc1.w, wf1T[i], 256, 1024);
    transpose_f16_kernel<<<eg(262144), 256, 0, stream>>>(blk[i].fc2.w, wf2T[i], 1024, 256);
  }
  for (int j = 0; j < 3; ++j)
    transpose_f16_kernel<<<eg(32768), 256, 0, stream>>>(pr[j].w1, wp1T[j], 256, 128);
  convpack_kernel<<<eg(9 * 64 * 32),   256, 0, stream>>>(c1w, wpk1, 64, 32);
  convpack_kernel<<<eg(9 * 128 * 64),  256, 0, stream>>>(c2w, wpk2, 128, 64);
  convpack_kernel<<<eg(9 * 256 * 128), 256, 0, stream>>>(c3w, wpk3, 256, 128);
  convpack_kernel<<<eg(9 * 256 * 256), 256, 0, stream>>>(rpew, wpkR, 256, 256);

  // ---- SPS ----
  conv0_kernel<<<eg((long long)8 * 16384 * 32), 256, 0, stream>>>(xin, c0w, conv_out);
  bn_lif_pad_kernel<<<eg((long long)2 * 16384 * 32), 256, 0, stream>>>(
      conv_out, bn0.g, bn0.b, bn0.m, bn0.v, s0, 16384, 128, 128, 32, 2 * 16384 * 32);
  conv3x3_wmma_kernel<<<dim3(64 / 32, 131072 / 128), 128, 0, stream>>>(s0, wpk1, conv_out, 128, 32, 64, 131072, 14, 7);
  bn_lif_pad_kernel<<<eg((long long)2 * 16384 * 64), 256, 0, stream>>>(
      conv_out, bn1.g, bn1.b, bn1.m, bn1.v, s1, 16384, 128, 128, 64, 2 * 16384 * 64);
  conv3x3_wmma_kernel<<<dim3(128 / 32, 131072 / 128), 128, 0, stream>>>(s1, wpk2, conv_out, 128, 64, 128, 131072, 14, 7);
  bn_lif_pad_kernel<<<eg((long long)2 * 16384 * 128), 256, 0, stream>>>(
      conv_out, bn2.g, bn2.b, bn2.m, bn2.v, s2, 16384, 128, 128, 128, 2 * 16384 * 128);
  pool_kernel<<<eg((long long)8 * 64 * 64 * 128), 256, 0, stream>>>(s2, s2p, 128, 128, 8 * 64 * 64 * 128);
  conv3x3_wmma_kernel<<<dim3(256 / 32, 32768 / 128), 128, 0, stream>>>(s2p, wpk3, conv_out, 64, 128, 256, 32768, 12, 6);
  bn_lif_pad_kernel<<<eg((long long)2 * 4096 * 256), 256, 0, stream>>>(
      conv_out, bn3.g, bn3.b, bn3.m, bn3.v, s3, 4096, 64, 64, 256, 2 * 4096 * 256);
  pool_kernel<<<eg((long long)8 * 32 * 32 * 256), 256, 0, stream>>>(s3, s3p, 64, 256, 8 * 32 * 32 * 256);
  conv3x3_wmma_kernel<<<dim3(256 / 32, 8192 / 128), 128, 0, stream>>>(s3p, wpkR, conv_out, 32, 256, 256, 8192, 10, 5);
  lif_tokens_kernel<<<eg((long long)2 * 1024 * 256), 256, 0, stream>>>(
      conv_out, rpebn.g, rpebn.b, rpebn.m, rpebn.v, s3p, xA, xhA);

  // ---- transformer blocks with static prune schedule ----
  static const int Ns[4]  = {1024, 716, 501, 351};
  static const int Nps[4] = {1024, 736, 512, 352};
  float* xcur = xA;  _Float16* xhcur = xhA;
  float* xoth = xB;  _Float16* xhoth = xhB;
  int N = 1024, Np = 1024;

  for (int i = 0; i < 4; ++i) {
    if (i >= 1) {
      int j = i - 1;
      int kNew = Ns[i], NpNew = Nps[i];
      mean_kernel<<<eg((long long)2 * N * 256), 256, 0, stream>>>(xcur, xm16, N, Np, 2 * N * 256);
      gemm(stream, xm16, 0, 256, wp1T[j], 0, 256, predh, 0, 128,
           pr[j].b1, 1.f, 2 * Np, 128, 256, 1, 0);
      pred_score_kernel<<<eg((long long)2 * N), 256, 0, stream>>>(predh, pr[j].w2, pr[j].b2, score, N, Np, 2 * N);
      topk_kernel<<<2, 256, 0, stream>>>(score, idxb, N, kNew);
      gather_kernel<<<eg((long long)8 * kNew * 256), 256, 0, stream>>>(
          xcur, idxb, xoth, xhoth, Np, NpNew, kNew, 8 * kNew * 256);
      { float* tf = xcur; xcur = xoth; xoth = tf; }
      { _Float16* th = xhcur; xhcur = xhoth; xhoth = th; }
      N = kNew; Np = NpNew;
    }
    const Blkp& bp = blk[i];
    int M = 8 * Np;
    // q / k / v linear -> BN -> LIF spikes (per-head layouts)
    gemm(stream, xhcur, 0, 256, wqT[i], 0, 256, lin_out, 0, 256, bp.q.bias, 1.f, M, 256, 256, 1, 0);
    lif_qkv_kernel<<<eg((long long)2 * N * 256), 256, 0, stream>>>(
        lin_out, bp.q.bn.g, bp.q.bn.b, bp.q.bn.m, bp.q.bn.v, qs, N, Np, 0, 2 * N * 256);
    gemm(stream, xhcur, 0, 256, wkT[i], 0, 256, lin_out, 0, 256, bp.k.bias, 1.f, M, 256, 256, 1, 0);
    lif_qkv_kernel<<<eg((long long)2 * N * 256), 256, 0, stream>>>(
        lin_out, bp.k.bn.g, bp.k.bn.b, bp.k.bn.m, bp.k.bn.v, ks, N, Np, 0, 2 * N * 256);
    gemm(stream, xhcur, 0, 256, wvT[i], 0, 256, lin_out, 0, 256, bp.v.bias, 1.f, M, 256, 256, 1, 0);
    lif_qkv_kernel<<<eg((long long)2 * N * 256), 256, 0, stream>>>(
        lin_out, bp.v.bn.g, bp.v.bn.b, bp.v.bn.m, bp.v.bn.v, vT, N, Np, 1, 2 * N * 256);
    // attention: S = q k^T * 0.125 (f16 out, exact), y = S v ; batch (b,h) per t
    for (int t = 0; t < 4; ++t) {
      const _Float16* qA = qs + (size_t)t * 2 * 8 * Np * 32;
      const _Float16* kB = ks + (size_t)t * 2 * 8 * Np * 32;
      gemm(stream, qA, (long long)Np * 32, 32, kB, (long long)Np * 32, 32,
           S16, (long long)Np * Np, Np, nullptr, 0.125f, Np, Np, 32, 16, 1);
      const _Float16* vB = vT + (size_t)t * 2 * 8 * 32 * Np;
      float* yC = yatt + (size_t)t * 2 * 8 * Np * 32;
      gemm(stream, S16, (long long)Np * Np, Np, vB, (long long)32 * Np, Np,
           yC, (long long)Np * 32, 32, nullptr, 1.f, Np, 32, Np, 16, 0);
    }
    lif_attn_kernel<<<eg((long long)2 * 8 * N * 32), 256, 0, stream>>>(yatt, ysp, N, Np, 2 * 8 * N * 32);
    // proj -> BN -> LIF -> residual
    gemm(stream, ysp, 0, 256, wpT[i], 0, 256, lin_out, 0, 256, bp.proj.bias, 1.f, M, 256, 256, 1, 0);
    lif_res_kernel<<<eg((long long)2 * N * 256), 256, 0, stream>>>(
        lin_out, bp.proj.bn.g, bp.proj.bn.b, bp.proj.bn.m, bp.proj.bn.v, xcur, xhcur, N, Np, 2 * N * 256);
    // MLP
    gemm(stream, xhcur, 0, 256, wf1T[i], 0, 256, lin_out, 0, 1024, bp.fc1.bias, 1.f, M, 1024, 256, 1, 0);
    lif_fc1_kernel<<<eg((long long)2 * N * 1024), 256, 0, stream>>>(
        lin_out, bp.fc1.bn.g, bp.fc1.bn.b, bp.fc1.bn.m, bp.fc1.bn.v, hsp, N, Np, 2 * N * 1024);
    gemm(stream, hsp, 0, 1024, wf2T[i], 0, 1024, lin_out, 0, 256, bp.fc2.bias, 1.f, M, 256, 1024, 1, 0);
    lif_res_kernel<<<eg((long long)2 * N * 256), 256, 0, stream>>>(
        lin_out, bp.fc2.bn.g, bp.fc2.bn.b, bp.fc2.bn.m, bp.fc2.bn.v, xcur, xhcur, N, Np, 2 * N * 256);
  }

  head_kernel<<<2, 256, 0, stream>>>(xcur, head_w, head_b, (float*)d_out, N, Np);
}
